// EGNNLayer_30820685316238
// MI455X (gfx1250) — matrix operand order repaired and checked by reading, not compiled, not run on hardware
//
#include <hip/hip_runtime.h>
#include <hip/hip_bf16.h>

typedef __attribute__((ext_vector_type(16))) _Float16 v16h;
typedef __attribute__((ext_vector_type(8)))  float    v8f;

#define HID    128
#define LDSTR  132   // padded LDS row stride (floats) to dodge bank conflicts

// ---------------------------------------------------------------------------
// WMMA fragment index helpers (wave32, 16x16x32 f16):
//  A (16x32): lane l -> row M = l&15, group g = l>>4.
//    half i (0..15): K = i + g*16
//  B (32x16): lane l -> col N = l&15, group g = l>>4.
//    half i (0..15): K = i + g*16
//  C/D (16x16 f32): lane l -> col N = l&15 ; vgpr r -> row M = r + 8*(l>>4)
// ---------------------------------------------------------------------------

__device__ __forceinline__ v16h wmma_load_a_f32row(const float* rowptr, int g) {
    v16h a;
#pragma unroll
    for (int v = 0; v < 8; ++v) {
        int k0 = g * 16 + v * 2;
        float x0 = rowptr[k0];
        float x1 = rowptr[k0 + 1];
        a[2 * v]     = (_Float16)x0;
        a[2 * v + 1] = (_Float16)x1;
    }
    return a;
}

__device__ __forceinline__ v8f wmma_mac(v16h a, v16h b, v8f c) {
    return __builtin_amdgcn_wmma_f32_16x16x32_f16(false, a, false, b, (short)0, c,
                                                  false, false);
}

__device__ __forceinline__ float silu_f(float x) {
    return x / (1.0f + __expf(-x));
}

// ---------------------------------------------------------------------------
// Pack a [K,128] row-major f32 weight matrix into WMMA-B fragment order, f16.
// Block b = c*8 + t handles K-chunk c (32 rows) x N-tile t (16 cols):
//   dst[b*512 + lane*16 + i] = W[c*32 + i + (lane>>4)*16][t*16 + (lane&15)]
// so each lane's 16 halves (32B) are contiguous -> single b128x2 load later.
// ---------------------------------------------------------------------------
__global__ void egnn_pack_wmma_b(const float* __restrict__ W,
                                 _Float16* __restrict__ dst) {
    int blk = blockIdx.x;            // c*8 + t
    int c = blk >> 3, t = blk & 7;
    for (int idx = threadIdx.x; idx < 512; idx += blockDim.x) {
        int lane = idx >> 4, i = idx & 15;
        int g = lane >> 4, n = lane & 15;
        int K = c * 32 + i + g * 16;
        dst[(size_t)blk * 512 + idx] = (_Float16)W[(size_t)K * HID + t * 16 + n];
    }
}

__global__ void egnn_zero_f32(float* __restrict__ p, long n) {
    long i = (long)blockIdx.x * blockDim.x + threadIdx.x;
    long stride = (long)gridDim.x * blockDim.x;
    for (; i < n; i += stride) p[i] = 0.0f;
}

// cooperative copy of packed weights into LDS (128 threads, 16B chunks)
__device__ __forceinline__ void stage_weights(uint4* dst, const uint4* src,
                                              int n16) {
    for (int i = threadIdx.x; i < n16; i += 128) dst[i] = src[i];
}

// ---------------------------------------------------------------------------
// Edge kernel: weights LDS-resident per block; each wave strides over
// 16-edge tiles.  stage1: [16x257] x mw1 -> SiLU (radial row = rank-1 fix);
// stage2: hidden x mw2 -> msg; coord mlp (VALU); native f32 atomic scatter.
// ---------------------------------------------------------------------------
__global__ __launch_bounds__(128) void egnn_edge_kernel(
    const float* __restrict__ h, const float* __restrict__ pos,
    const long long* __restrict__ ei,
    const _Float16* __restrict__ mw1p, const float* __restrict__ mw1_last,
    const float* __restrict__ mb1, const _Float16* __restrict__ mw2p,
    const float* __restrict__ mb2,
    const float* __restrict__ cw1, const float* __restrict__ cb1,
    const float* __restrict__ cw2, const float* __restrict__ cb2,
    float* __restrict__ agg_msg, float* __restrict__ agg_trans, int E) {
    __shared__ __align__(32) _Float16 s_w1[8 * 8 * 512];  // 64 KB mw1 fragments
    __shared__ __align__(32) _Float16 s_w2[4 * 8 * 512];  // 32 KB mw2 fragments
    __shared__ float s_hid[4][16 * LDSTR];                // ~33 KB msg staging
    __shared__ float s_rad[4][16];
    __shared__ int   s_row[4][16];
    __shared__ int   s_col[4][16];
    __shared__ float s_cd[4][3][16];

    stage_weights((uint4*)s_w1, (const uint4*)mw1p, 8 * 8 * 512 / 8);
    stage_weights((uint4*)s_w2, (const uint4*)mw2p, 4 * 8 * 512 / 8);
    __syncthreads();

    const int wave = threadIdx.x >> 5;
    const int lane = threadIdx.x & 31;
    const int g = lane >> 4, n = lane & 15;
    const int ntiles = (E + 15) / 16;

    for (int tile = blockIdx.x * 4 + wave; tile < ntiles; tile += gridDim.x * 4) {
        const int ebase = tile * 16;

        // --- edge setup (lanes 0..15), wavefront-synchronous LDS ---------
        if (lane < 16) {
            int e = ebase + lane;
            int r = 0, c = 0;
            float dx = 0.f, dy = 0.f, dz = 0.f, rad = 0.f;
            if (e < E) {
                r = (int)ei[e];
                c = (int)ei[(size_t)E + e];
                dx = pos[r * 3 + 0] - pos[c * 3 + 0];
                dy = pos[r * 3 + 1] - pos[c * 3 + 1];
                dz = pos[r * 3 + 2] - pos[c * 3 + 2];
                rad = dx * dx + dy * dy + dz * dz + 1e-8f;
            }
            s_row[wave][lane] = r;
            s_col[wave][lane] = c;
            s_rad[wave][lane] = rad;
            s_cd[wave][0][lane] = dx;
            s_cd[wave][1][lane] = dy;
            s_cd[wave][2][lane] = dz;
        }

        const int rowA = s_row[wave][n];
        const int colA = s_col[wave][n];

        // --- stage 1: K = 256 (h[row] ‖ h[col]) --------------------------
        v8f acc[8];
#pragma unroll
        for (int t = 0; t < 8; ++t) acc[t] = (v8f){};

        for (int c4 = 0; c4 < 8; ++c4) {
            const float* src = (c4 < 4)
                                   ? (h + (size_t)rowA * HID + c4 * 32)
                                   : (h + (size_t)colA * HID + (c4 - 4) * 32);
            v16h a = wmma_load_a_f32row(src, g);
#pragma unroll
            for (int t = 0; t < 8; ++t) {
                v16h b = *(const v16h*)(&s_w1[(c4 * 8 + t) * 512 + lane * 16]);
                acc[t] = wmma_mac(a, b, acc[t]);
            }
        }
        // radial rank-1 term + bias + SiLU, stage hidden into LDS
#pragma unroll
        for (int t = 0; t < 8; ++t) {
            float wl = mw1_last[t * 16 + n];
            float bi = mb1[t * 16 + n];
#pragma unroll
            for (int r = 0; r < 8; ++r) {
                int M = r + 8 * g;
                float x = acc[t][r] + s_rad[wave][M] * wl + bi;
                s_hid[wave][M * LDSTR + t * 16 + n] = silu_f(x);
            }
            acc[t] = (v8f){};  // reuse accumulator for stage 2
        }

        // --- stage 2: K = 128 (hidden from LDS) --------------------------
        for (int c2 = 0; c2 < 4; ++c2) {
            v16h a = wmma_load_a_f32row(&s_hid[wave][n * LDSTR + c2 * 32], g);
#pragma unroll
            for (int t = 0; t < 8; ++t) {
                v16h b = *(const v16h*)(&s_w2[(c2 * 8 + t) * 512 + lane * 16]);
                acc[t] = wmma_mac(a, b, acc[t]);
            }
        }
        // msg = acc + mb2  -> back into LDS (overwrites hidden)
#pragma unroll
        for (int t = 0; t < 8; ++t) {
            float bi = mb2[t * 16 + n];
#pragma unroll
            for (int r = 0; r < 8; ++r) {
                int M = r + 8 * g;
                s_hid[wave][M * LDSTR + t * 16 + n] = acc[t][r] + bi;
            }
        }

        // --- coord mlp + trans scatter (lanes 0..15, one edge each) ------
        if (lane < 16) {
            int e = ebase + lane;
            if (e < E) {
                float sj0 = cb1[0], sj1 = cb1[1], sj2 = cb1[2], sj3 = cb1[3];
                const float* m = &s_hid[wave][lane * LDSTR];
                for (int k = 0; k < HID; ++k) {
                    float mv = m[k];
                    sj0 += mv * cw1[k * 4 + 0];
                    sj1 += mv * cw1[k * 4 + 1];
                    sj2 += mv * cw1[k * 4 + 2];
                    sj3 += mv * cw1[k * 4 + 3];
                }
                float cw = cb2[0] + silu_f(sj0) * cw2[0] + silu_f(sj1) * cw2[1] +
                           silu_f(sj2) * cw2[2] + silu_f(sj3) * cw2[3];
                int r = s_row[wave][lane];
                unsafeAtomicAdd(&agg_trans[r * 3 + 0], s_cd[wave][0][lane] * cw);
                unsafeAtomicAdd(&agg_trans[r * 3 + 1], s_cd[wave][1][lane] * cw);
                unsafeAtomicAdd(&agg_trans[r * 3 + 2], s_cd[wave][2][lane] * cw);
            }
        }

        // --- agg_msg scatter: whole wave per edge, coalesced columns -----
        for (int m = 0; m < 16; ++m) {
            if (ebase + m >= E) break;
            int r = s_row[wave][m];
            float* dst = agg_msg + (size_t)r * HID;
            const float* src = &s_hid[wave][m * LDSTR];
#pragma unroll
            for (int cc = 0; cc < 4; ++cc)
                unsafeAtomicAdd(&dst[cc * 32 + lane], src[cc * 32 + lane]);
        }
    }
}

// ---------------------------------------------------------------------------
// Node kernel: h_new = h + MLP([h ‖ agg_msg]); weights LDS-resident,
// node-tile stride loop, 16 nodes per wave.
// ---------------------------------------------------------------------------
__global__ __launch_bounds__(128) void egnn_node_kernel(
    const float* __restrict__ h, const float* __restrict__ agg_msg,
    const _Float16* __restrict__ nw1p, const float* __restrict__ nb1,
    const _Float16* __restrict__ nw2p, const float* __restrict__ nb2,
    float* __restrict__ hout, int N) {
    __shared__ __align__(32) _Float16 s_w1[8 * 8 * 512];  // 64 KB nw1 fragments
    __shared__ __align__(32) _Float16 s_w2[4 * 8 * 512];  // 32 KB nw2 fragments
    __shared__ float s_hid[4][16 * LDSTR];

    stage_weights((uint4*)s_w1, (const uint4*)nw1p, 8 * 8 * 512 / 8);
    stage_weights((uint4*)s_w2, (const uint4*)nw2p, 4 * 8 * 512 / 8);
    __syncthreads();

    const int wave = threadIdx.x >> 5;
    const int lane = threadIdx.x & 31;
    const int g = lane >> 4, n = lane & 15;
    const int ntiles = (N + 15) / 16;

    for (int tile = blockIdx.x * 4 + wave; tile < ntiles; tile += gridDim.x * 4) {
        const int base = tile * 16;
        int node = base + n;
        int nodeC = node < N ? node : (N - 1);  // clamp gathers, guard stores

        v8f acc[8];
#pragma unroll
        for (int t = 0; t < 8; ++t) acc[t] = (v8f){};

        for (int c4 = 0; c4 < 8; ++c4) {
            const float* src =
                (c4 < 4) ? (h + (size_t)nodeC * HID + c4 * 32)
                         : (agg_msg + (size_t)nodeC * HID + (c4 - 4) * 32);
            v16h a = wmma_load_a_f32row(src, g);
#pragma unroll
            for (int t = 0; t < 8; ++t) {
                v16h b = *(const v16h*)(&s_w1[(c4 * 8 + t) * 512 + lane * 16]);
                acc[t] = wmma_mac(a, b, acc[t]);
            }
        }
#pragma unroll
        for (int t = 0; t < 8; ++t) {
            float bi = nb1[t * 16 + n];
#pragma unroll
            for (int r = 0; r < 8; ++r) {
                int M = r + 8 * g;
                s_hid[wave][M * LDSTR + t * 16 + n] = silu_f(acc[t][r] + bi);
            }
            acc[t] = (v8f){};
        }

        for (int c2 = 0; c2 < 4; ++c2) {
            v16h a = wmma_load_a_f32row(&s_hid[wave][n * LDSTR + c2 * 32], g);
#pragma unroll
            for (int t = 0; t < 8; ++t) {
                v16h b = *(const v16h*)(&s_w2[(c2 * 8 + t) * 512 + lane * 16]);
                acc[t] = wmma_mac(a, b, acc[t]);
            }
        }
#pragma unroll
        for (int t = 0; t < 8; ++t) {
            float bi = nb2[t * 16 + n];
#pragma unroll
            for (int r = 0; r < 8; ++r) {
                int M = r + 8 * g;
                s_hid[wave][M * LDSTR + t * 16 + n] = acc[t][r] + bi;
            }
        }

        // residual add + coalesced store
        for (int m = 0; m < 16; ++m) {
            int nd = base + m;
            if (nd >= N) break;
            const float* src = &s_hid[wave][m * LDSTR];
            float* dst = hout + (size_t)nd * HID;
            const float* hrow = h + (size_t)nd * HID;
#pragma unroll
            for (int cc = 0; cc < 4; ++cc)
                dst[cc * 32 + lane] = hrow[cc * 32 + lane] + src[cc * 32 + lane];
        }
    }
}

__global__ void egnn_pos_kernel(const float* __restrict__ pos,
                                const float* __restrict__ agg_trans,
                                float* __restrict__ pos_out, int n3) {
    int i = blockIdx.x * blockDim.x + threadIdx.x;
    if (i < n3) pos_out[i] = pos[i] + agg_trans[i];
}

// ---------------------------------------------------------------------------
extern "C" void kernel_launch(void* const* d_in, const int* in_sizes, int n_in,
                              void* d_out, int out_size, void* d_ws, size_t ws_size,
                              hipStream_t stream) {
    const float*     h    = (const float*)d_in[0];
    const float*     pos  = (const float*)d_in[1];
    const long long* ei   = (const long long*)d_in[2];  // int64 edge_index [2,E]
    const float*     mw1  = (const float*)d_in[3];      // [257,128]
    const float*     mb1  = (const float*)d_in[4];
    const float*     mw2  = (const float*)d_in[5];      // [128,128]
    const float*     mb2  = (const float*)d_in[6];
    const float*     cw1  = (const float*)d_in[7];      // [128,4]
    const float*     cb1  = (const float*)d_in[8];
    const float*     cw2  = (const float*)d_in[9];      // [4,1]
    const float*     cb2  = (const float*)d_in[10];
    const float*     nw1  = (const float*)d_in[11];     // [256,128]
    const float*     nb1  = (const float*)d_in[12];
    const float*     nw2  = (const float*)d_in[13];     // [128,128]
    const float*     nb2  = (const float*)d_in[14];

    const int N = in_sizes[0] / HID;   // 50000
    const int E = in_sizes[2] / 2;     // 600000

    // workspace layout
    char* ws = (char*)d_ws;
    _Float16* mw1p = (_Float16*)(ws + 0);            //  64 KB (8 chunks x 8 tiles)
    _Float16* mw2p = (_Float16*)(ws + 65536);        //  32 KB
    _Float16* nw1p = (_Float16*)(ws + 98304);        //  64 KB
    _Float16* nw2p = (_Float16*)(ws + 163840);       //  32 KB
    float*    agg_msg   = (float*)(ws + 196608);     // N*128*4 = 25.6 MB
    float*    agg_trans = agg_msg + (size_t)N * HID; // N*3*4
    const float* mw1_last = mw1 + (size_t)256 * HID; // radial row of mw1

    // 1) pack weights into WMMA-B fragment layout (f16)
    egnn_pack_wmma_b<<<64, 128, 0, stream>>>(mw1, mw1p);  // K=256
    egnn_pack_wmma_b<<<32, 128, 0, stream>>>(mw2, mw2p);  // K=128
    egnn_pack_wmma_b<<<64, 128, 0, stream>>>(nw1, nw1p);  // K=256
    egnn_pack_wmma_b<<<32, 128, 0, stream>>>(nw2, nw2p);  // K=128

    // 2) zero accumulators (agg_msg ‖ agg_trans contiguous)
    long nzero = (long)N * HID + (long)N * 3;
    egnn_zero_f32<<<2048, 256, 0, stream>>>(agg_msg, nzero);

    // 3) edge pass (message MLP + coord MLP + segment sums)
    egnn_edge_kernel<<<1024, 128, 0, stream>>>(
        h, pos, ei, mw1p, mw1_last, mb1, mw2p, mb2, cw1, cb1, cw2, cb2,
        agg_msg, agg_trans, E);

    // 4) node pass (node MLP + residual), then positions
    float* h_out = (float*)d_out;                   // [N,128]
    float* pos_out = h_out + (size_t)N * HID;       // [N,3]
    egnn_node_kernel<<<256, 128, 0, stream>>>(h, agg_msg, nw1p, nb1, nw2p,
                                              nb2, h_out, N);
    egnn_pos_kernel<<<(N * 3 + 255) / 256, 256, 0, stream>>>(pos, agg_trans,
                                                             pos_out, N * 3);
}
